// AirConv3D_10840497455735
// MI455X (gfx1250) — compile-verified, matrix operand order checked by reading, not run on hardware
//
#include <hip/hip_runtime.h>
#include <hip/hip_bf16.h>
#include <stdint.h>

// ---------------------------------------------------------------------------
// CP-decomposed 3D conv, fused for MI455X (gfx1250, wave32, WMMA, async-LDS).
//
//   prep:   x[c][n] fp32 -> Xb[n][c] bf16; pack bf16 weight blocks; zero page
//   fused:  per 4x4x8 voxel tile:
//     S1: async halo load  Xb(6x6x10 cols x 32ch) -> LDS   (ASYNCcnt path)
//     S2: WMMA GEMM-in  (M=64pad-rank, K=32)  LDS -> LDS
//     S3-5: separable 3-tap convs H/W/D in LDS (weights in registers,
//           rank-pair per thread, packed 32-bit LDS traffic)
//     S6: WMMA GEMM-out (M=64 cout, K=64) + LDS bias -> global fp32 (NT)
// ---------------------------------------------------------------------------

#define RANK   53
#define CIN    32
#define COUT   64
#define RPAD   64
#define NVOX   (56 * 56 * 56)      // 175616

// tile geometry
#define TBH 4
#define TBW 4
#define TBD 8
#define HHL 6
#define HWL 6
#define HDL 10
#define XCOLS      (HHL * HWL * HDL)   // 360 halo columns
#define XCOLS_PAD  368                 // 23 WMMA n-tiles
#define NT_IN      (XCOLS_PAD / 16)    // 23
#define CH_COLS    (TBH * HWL * HDL)   // 240
#define CW_COLS    (TBH * TBW * HDL)   // 160
#define CD_COLS    (TBH * TBW * TBD)   // 128

// LDS layout (bytes)
#define XH_OFF   0
#define T1H_OFF  (XH_OFF  + XCOLS_PAD * CIN  * 2)   // 23552
#define CH_OFF   (T1H_OFF + XCOLS_PAD * RPAD * 2)   // 70656
#define CW_OFF   (CH_OFF  + CH_COLS   * RPAD * 2)   // 101376
#define CD_OFF   (CW_OFF  + CW_COLS   * RPAD * 2)   // 121856
#define UW_OFF   (CD_OFF  + CD_COLS   * RPAD * 2)   // 138240
#define BIAS_OFF (UW_OFF  + 3 * 3 * RPAD * 4)       // 140544
#define SMEM_BYTES (BIAS_OFF + COUT * 4)            // 140800

typedef __attribute__((ext_vector_type(8)))  __bf16          v8bf;
typedef __attribute__((ext_vector_type(16))) __bf16          v16bf;
typedef __attribute__((ext_vector_type(8)))  float           v8f;
typedef __attribute__((ext_vector_type(4)))  unsigned int    v4u;

// exact RNE scalar (cold path: weight packing only)
__device__ __forceinline__ unsigned short f2bf(float f) {
    union { float f; unsigned int u; } v; v.f = f;
    unsigned int u = v.u;
    return (unsigned short)((u + 0x7FFFu + ((u >> 16) & 1u)) >> 16);
}
// hot path: two fp32 -> packed bf16x2.  Round-half-up (+0x8000) then one
// v_perm_b32 grabs both high halves: 3 VALU total, short dependency chain.
__device__ __forceinline__ unsigned int pack2_bf16(float lo, float hi) {
#if __has_builtin(__builtin_amdgcn_cvt_pk_bf16_f32)
    typedef __attribute__((ext_vector_type(2))) __bf16 v2bf;
    v2bf p = __builtin_amdgcn_cvt_pk_bf16_f32(lo, hi);
    return __builtin_bit_cast(unsigned int, p);
#else
    unsigned int ulo = __builtin_bit_cast(unsigned int, lo) + 0x8000u;
    unsigned int uhi = __builtin_bit_cast(unsigned int, hi) + 0x8000u;
    return __builtin_amdgcn_perm(uhi, ulo, 0x07060302u);  // {uhi.hi16, ulo.hi16}
#endif
}
__device__ __forceinline__ float blo(unsigned int u) {
    union { unsigned int u; float f; } v; v.u = u << 16;            return v.f;
}
__device__ __forceinline__ float bhi(unsigned int u) {
    union { unsigned int u; float f; } v; v.u = u & 0xFFFF0000u;    return v.f;
}
__device__ __forceinline__ v8bf ld8(const unsigned short* p) {
    return *(const v8bf*)p;   // 16B-aligned (global or LDS)
}
__device__ __forceinline__ v16bf cat16(v8bf lo, v8bf hi) {
    return __builtin_shufflevector(lo, hi, 0,1,2,3,4,5,6,7,8,9,10,11,12,13,14,15);
}

// ---------------------------------------------------------------------------
// prep: W1[r][c]=U_cin[c][r] (64x32), W2[o][r]=U_cout[r][o] (64x64), zero page
// ---------------------------------------------------------------------------
__global__ void prep_w_kernel(const float* __restrict__ U_cin,
                              const float* __restrict__ U_cout,
                              unsigned short* __restrict__ W1,
                              unsigned short* __restrict__ W2,
                              unsigned short* __restrict__ ZP) {
    int tid = threadIdx.x;
    for (int i = tid; i < RPAD * CIN; i += blockDim.x) {
        int r = i / CIN, c = i % CIN;
        W1[i] = f2bf(r < RANK ? U_cin[c * RANK + r] : 0.0f);
    }
    for (int i = tid; i < COUT * RPAD; i += blockDim.x) {
        int o = i / RPAD, r = i % RPAD;
        W2[i] = f2bf(r < RANK ? U_cout[r * COUT + o] : 0.0f);
    }
    for (int i = tid; i < 128; i += blockDim.x) ZP[i] = 0;
}

// x[c][n] fp32 -> Xb[n][c] bf16, two channels packed per 32-bit store
__global__ void prep_x_kernel(const float* __restrict__ x,
                              unsigned short* __restrict__ Xb) {
    size_t idx = (size_t)blockIdx.x * blockDim.x + threadIdx.x;
    if (idx >= (size_t)CIN * NVOX / 2) return;
    size_t base = idx * 2;
    int    c = (int)(base & (CIN - 1));
    size_t n = base >> 5;
    float lo = __builtin_nontemporal_load(&x[(size_t)c       * NVOX + n]);
    float hi = __builtin_nontemporal_load(&x[(size_t)(c + 1) * NVOX + n]);
    *(unsigned int*)(Xb + base) = pack2_bf16(lo, hi);
}

// ---------------------------------------------------------------------------
// Fused tile kernel
// ---------------------------------------------------------------------------
__global__ void __launch_bounds__(256)
fused_kernel(const unsigned short* __restrict__ Xb,   // [NVOX][32] bf16
             const unsigned short* __restrict__ W1,   // [64][32]  bf16
             const unsigned short* __restrict__ W2,   // [64][64]  bf16
             const float* __restrict__ U_kh,          // (3,53)
             const float* __restrict__ U_kw,
             const float* __restrict__ U_kd,
             const float* __restrict__ bias,          // (64)
             const unsigned short* __restrict__ ZP,   // zero page
             float* __restrict__ Y) {                 // [64][NVOX] fp32
    extern __shared__ char smem[];
    unsigned short* XH  = (unsigned short*)(smem + XH_OFF);
    unsigned short* T1H = (unsigned short*)(smem + T1H_OFF);
    unsigned short* CH  = (unsigned short*)(smem + CH_OFF);
    unsigned short* CW  = (unsigned short*)(smem + CW_OFF);
    unsigned short* CD  = (unsigned short*)(smem + CD_OFF);
    float*          UW  = (float*)(smem + UW_OFF);    // [3 axes][3 taps][64]
    float*          BS  = (float*)(smem + BIAS_OFF);  // [64]

    const int tid  = threadIdx.x;
    const int lane = tid & 31;
    const int wv   = tid >> 5;
    const int nc   = lane & 15;        // WMMA column / M-row within tile
    const int hq   = lane >> 4;        // K-half selector

    const int d0 = blockIdx.x * TBD;
    const int w0 = blockIdx.y * TBW;
    const int h0 = blockIdx.z * TBH;

    // --- S1: async halo copy global -> LDS (OOB columns read the zero page) -
    for (int i = tid; i < XCOLS_PAD * 4; i += 256) {
        int col = i >> 2, part = i & 3;                 // 4 x 16B per column
        const unsigned short* src = ZP;
        if (col < XCOLS) {
            int dd = col % HDL, rest = col / HDL;
            int ww = rest % HWL, hh = rest / HWL;
            int gh = h0 - 1 + hh, gw = w0 - 1 + ww, gd = d0 - 1 + dd;
            if ((unsigned)gh < 56u && (unsigned)gw < 56u && (unsigned)gd < 56u) {
                size_t n = ((size_t)gh * 56 + gw) * 56 + gd;
                src = Xb + n * CIN + part * 8;
            }
        }
        unsigned lds_addr = (unsigned)(size_t)(XH + col * CIN + part * 8);
        unsigned long long gaddr = (unsigned long long)(size_t)src;
        asm volatile("global_load_async_to_lds_b128 %0, %1, off"
                     :: "v"(lds_addr), "v"(gaddr) : "memory");
    }

    // overlap: conv weights + bias into LDS (ranks zero-padded to 64)
    for (int i = tid; i < 3 * 3 * RPAD; i += 256) {
        int axis = i / (3 * RPAD);
        int rem  = i % (3 * RPAD);
        int tap  = rem / RPAD;
        int r    = rem % RPAD;
        const float* U = (axis == 0) ? U_kh : (axis == 1) ? U_kw : U_kd;
        UW[i] = (r < RANK) ? U[tap * RANK + r] : 0.0f;
    }
    if (tid < COUT) BS[tid] = bias[tid];

    // overlap: preload GEMM-in A-fragments (invariant across tiles)
    v16bf a_in[4];
#pragma unroll
    for (int mt = 0; mt < 4; ++mt) {
        const unsigned short* arow = W1 + (size_t)(mt * 16 + nc) * CIN;
        a_in[mt] = cat16(ld8(arow + 8 * hq), ld8(arow + 16 + 8 * hq));
    }

    asm volatile("s_wait_asynccnt 0x0" ::: "memory");
    __syncthreads();

    // --- S2: GEMM-in on halo columns: T1H[col][r] = sum_c W1[r][c]*XH[col][c]
    for (int t = wv; t < NT_IN; t += 8) {              // wave-uniform loop
        const unsigned short* bcol = XH + (size_t)(t * 16 + nc) * CIN;
        v16bf b = cat16(ld8(bcol + 8 * hq), ld8(bcol + 16 + 8 * hq));
#pragma unroll
        for (int mt = 0; mt < 4; ++mt) {
            v8f c = {};
            c = __builtin_amdgcn_wmma_f32_16x16x32_bf16(
                    false, a_in[mt], false, b, (short)0, c, false, false);
            v4u pk;
            pk.x = pack2_bf16(c[0], c[1]);
            pk.y = pack2_bf16(c[2], c[3]);
            pk.z = pack2_bf16(c[4], c[5]);
            pk.w = pack2_bf16(c[6], c[7]);
            *(v4u*)(T1H + (size_t)(t * 16 + nc) * RPAD + mt * 16 + 8 * hq) = pk;
        }
    }
    __syncthreads();

    // each thread owns a fixed rank pair for all conv stages
    const int rr = (tid & 31) * 2;          // rank pair base (0..62)
    const int cg = tid >> 5;                // column group 0..7

    // --- S3: conv along H: CH[(h,ww,dd)] = sum_i UW_h[i] * T1H[(h+i,ww,dd)]
    {
        float u0l = UW[0 * RPAD + rr], u0h = UW[0 * RPAD + rr + 1];
        float u1l = UW[1 * RPAD + rr], u1h = UW[1 * RPAD + rr + 1];
        float u2l = UW[2 * RPAD + rr], u2h = UW[2 * RPAD + rr + 1];
        for (int col = cg; col < CH_COLS; col += 8) {
            int dd = col % HDL, rest = col / HDL;
            int ww = rest % HWL, h = rest / HWL;
            int sb = ww * HDL + dd;
            unsigned a0 = *(const unsigned*)(T1H + (size_t)((h    ) * 60 + sb) * RPAD + rr);
            unsigned a1 = *(const unsigned*)(T1H + (size_t)((h + 1) * 60 + sb) * RPAD + rr);
            unsigned a2 = *(const unsigned*)(T1H + (size_t)((h + 2) * 60 + sb) * RPAD + rr);
            float accl = u0l * blo(a0) + u1l * blo(a1) + u2l * blo(a2);
            float acch = u0h * bhi(a0) + u1h * bhi(a1) + u2h * bhi(a2);
            *(unsigned*)(CH + (size_t)col * RPAD + rr) = pack2_bf16(accl, acch);
        }
    }
    __syncthreads();

    // --- S4: conv along W
    {
        float u0l = UW[(3 + 0) * RPAD + rr], u0h = UW[(3 + 0) * RPAD + rr + 1];
        float u1l = UW[(3 + 1) * RPAD + rr], u1h = UW[(3 + 1) * RPAD + rr + 1];
        float u2l = UW[(3 + 2) * RPAD + rr], u2h = UW[(3 + 2) * RPAD + rr + 1];
        for (int col = cg; col < CW_COLS; col += 8) {
            int dd = col % HDL, rest = col / HDL;
            int w = rest % TBW, h = rest / TBW;
            unsigned a0 = *(const unsigned*)(CH + (size_t)((h * HWL + w    ) * HDL + dd) * RPAD + rr);
            unsigned a1 = *(const unsigned*)(CH + (size_t)((h * HWL + w + 1) * HDL + dd) * RPAD + rr);
            unsigned a2 = *(const unsigned*)(CH + (size_t)((h * HWL + w + 2) * HDL + dd) * RPAD + rr);
            float accl = u0l * blo(a0) + u1l * blo(a1) + u2l * blo(a2);
            float acch = u0h * bhi(a0) + u1h * bhi(a1) + u2h * bhi(a2);
            *(unsigned*)(CW + (size_t)col * RPAD + rr) = pack2_bf16(accl, acch);
        }
    }
    __syncthreads();

    // --- S5: conv along D
    {
        float u0l = UW[(6 + 0) * RPAD + rr], u0h = UW[(6 + 0) * RPAD + rr + 1];
        float u1l = UW[(6 + 1) * RPAD + rr], u1h = UW[(6 + 1) * RPAD + rr + 1];
        float u2l = UW[(6 + 2) * RPAD + rr], u2h = UW[(6 + 2) * RPAD + rr + 1];
        for (int col = cg; col < CD_COLS; col += 8) {
            int d = col & (TBD - 1), hw = col >> 3;
            unsigned a0 = *(const unsigned*)(CW + (size_t)(hw * HDL + d    ) * RPAD + rr);
            unsigned a1 = *(const unsigned*)(CW + (size_t)(hw * HDL + d + 1) * RPAD + rr);
            unsigned a2 = *(const unsigned*)(CW + (size_t)(hw * HDL + d + 2) * RPAD + rr);
            float accl = u0l * blo(a0) + u1l * blo(a1) + u2l * blo(a2);
            float acch = u0h * bhi(a0) + u1h * bhi(a1) + u2h * bhi(a2);
            *(unsigned*)(CD + (size_t)col * RPAD + rr) = pack2_bf16(accl, acch);
        }
    }
    __syncthreads();

    // --- S6: GEMM-out: Y[o][n] = sum_r W2[o][r] * CD[col][r] + bias[o] ------
    {
        const int t = wv;                               // 8 waves, 8 n-tiles
        const unsigned short* bcol = CD + (size_t)(t * 16 + nc) * RPAD;
        v16bf b0 = cat16(ld8(bcol +      8 * hq), ld8(bcol + 16 + 8 * hq));
        v16bf b1 = cat16(ld8(bcol + 32 + 8 * hq), ld8(bcol + 48 + 8 * hq));

        int l  = t * 16 + nc;                           // tile-local column
        int d  = l & 7, hw = l >> 3;
        int w  = hw & 3, h = hw >> 2;
        size_t n = ((size_t)(h0 + h) * 56 + (w0 + w)) * 56 + (d0 + d);

#pragma unroll
        for (int mt = 0; mt < 4; ++mt) {
            const unsigned short* arow = W2 + (size_t)(mt * 16 + nc) * RPAD;
            v16bf a0 = cat16(ld8(arow +      8 * hq), ld8(arow + 16 + 8 * hq));
            v16bf a1 = cat16(ld8(arow + 32 + 8 * hq), ld8(arow + 48 + 8 * hq));
            v8f c = {};
            c = __builtin_amdgcn_wmma_f32_16x16x32_bf16(
                    false, a0, false, b0, (short)0, c, false, false);
            c = __builtin_amdgcn_wmma_f32_16x16x32_bf16(
                    false, a1, false, b1, (short)0, c, false, false);
#pragma unroll
            for (int vg = 0; vg < 8; ++vg) {
                int o = mt * 16 + 8 * hq + vg;
                __builtin_nontemporal_store(c[vg] + BS[o], &Y[(size_t)o * NVOX + n]);
            }
        }
    }
}

// ---------------------------------------------------------------------------
// Host-side launch
// ---------------------------------------------------------------------------
extern "C" void kernel_launch(void* const* d_in, const int* in_sizes, int n_in,
                              void* d_out, int out_size, void* d_ws, size_t ws_size,
                              hipStream_t stream) {
    const float* x      = (const float*)d_in[0];
    const float* U_kh   = (const float*)d_in[1];
    const float* U_kw   = (const float*)d_in[2];
    const float* U_kd   = (const float*)d_in[3];
    const float* U_cin  = (const float*)d_in[4];
    const float* U_cout = (const float*)d_in[5];
    const float* bias   = (const float*)d_in[6];
    float* out = (float*)d_out;

    const size_t XB_BYTES = (size_t)CIN * NVOX * 2;     // 11,239,424
    char* ws = (char*)d_ws;
    unsigned short* Xb = (unsigned short*)(ws);
    unsigned short* W1 = (unsigned short*)(ws + XB_BYTES);
    unsigned short* W2 = (unsigned short*)(ws + XB_BYTES + RPAD * CIN * 2);
    unsigned short* ZP = (unsigned short*)(ws + XB_BYTES + RPAD * CIN * 2 + COUT * RPAD * 2);

    const int BLK = 256;  // 8 wave32s
    const int prep_blocks = (int)(((size_t)CIN * NVOX / 2 + BLK - 1) / BLK); // 10976
    dim3 grid(56 / TBD, 56 / TBW, 56 / TBH);                                 // 7x14x14

    prep_w_kernel<<<1,           BLK, 0, stream>>>(U_cin, U_cout, W1, W2, ZP);
    prep_x_kernel<<<prep_blocks, BLK, 0, stream>>>(x, Xb);
    fused_kernel <<<grid, BLK, SMEM_BYTES, stream>>>(Xb, W1, W2,
                                                     U_kh, U_kw, U_kd,
                                                     bias, ZP, out);
}
